// WindowAttention_15461882265852
// MI455X (gfx1250) — matrix operand order repaired and checked by reading, not compiled, not run on hardware
//
#include <hip/hip_runtime.h>

typedef _Float16 h16;
typedef __attribute__((ext_vector_type(16))) _Float16 v16h;
typedef __attribute__((ext_vector_type(8)))  _Float16 v8h;
typedef __attribute__((ext_vector_type(4)))  _Float16 v4h;
typedef __attribute__((ext_vector_type(8)))  float    v8f;

namespace {
constexpr int kC      = 256;   // model dim
constexpr int kW      = 128;   // window
constexpr int kKT     = 8;     // 256/32 K-chunks for C-dim reductions
constexpr int kNTqkv  = 48;    // 768/16 N-tiles of QKV gemm
constexpr int kNTproj = 16;    // 256/16 N-tiles of proj gemm

// LDS layout (in halves). All strides multiples of 8 halves (16B aligned rows).
constexpr int XS   = 264;            // x/out row stride (256 + 8 pad)
constexpr int XSZ  = 128 * XS;       // 33792
constexpr int QKS  = 72;             // per-head q|k row stride (64 + 8 pad)
constexpr int QKSZ = 128 * QKS;      // 9216
constexpr int VTS  = 136;            // v^T / P row stride (128 + 8 pad)
constexpr int VTSZ = 32 * VTS;       // 4352
constexpr int WSZ  = QKSZ + VTSZ;    // 13568 per wave
constexpr int LDS_HALVES = XSZ + 8 * WSZ;   // 142336 halves = 284672 B <= 320KB
}

__device__ __forceinline__ v16h ld_frag2(const h16* p0, const h16* p1) {
  union { v16h v; v8h h[2]; } u;
  u.h[0] = *(const v8h*)p0;
  u.h[1] = *(const v8h*)p1;
  return u.v;
}

// ---------------------------------------------------------------------------
// Kernel 1: repack f32 weights into f16 B-fragment order.
// Fragment (kt, nt): lane L supplies 16 halves j=0..15 with
//   K = kt*32 + (L>>4)*16 + j,  N = nt*16 + (L&15),  value = W[N][K]  (y = x W^T)
// Stored flat: ((kt*NT + nt)*32 + lane)*16 + j  -> 32B contiguous per lane.
// ---------------------------------------------------------------------------
__global__ void pack_weights_k(const float* __restrict__ qkv_w,
                               const float* __restrict__ proj_w,
                               h16* __restrict__ wq, h16* __restrict__ wp) {
  const int t = blockIdx.x * 256 + threadIdx.x;   // 0 .. 16383
  const int NQ = kKT * kNTqkv * 32;               // 12288
  if (t < NQ) {
    const int kt = t / (kNTqkv * 32);
    const int rem = t % (kNTqkv * 32);
    const int nt = rem >> 5, lane = rem & 31;
    const int n  = nt * 16 + (lane & 15);
    const int kb = kt * 32 + (lane >> 4) * 16;
    const float* src = qkv_w + n * kC + kb;
    h16* dst = wq + (size_t)t * 16;
#pragma unroll
    for (int j = 0; j < 16; ++j) dst[j] = (h16)src[j];
  } else {
    const int t2 = t - NQ;                         // 0 .. 4095
    const int kt = t2 / (kNTproj * 32);
    const int rem = t2 % (kNTproj * 32);
    const int nt = rem >> 5, lane = rem & 31;
    const int n  = nt * 16 + (lane & 15);
    const int kb = kt * 32 + (lane >> 4) * 16;
    const float* src = proj_w + n * kC + kb;
    h16* dst = wp + (size_t)t2 * 16;
#pragma unroll
    for (int j = 0; j < 16; ++j) dst[j] = (h16)src[j];
  }
}

// ---------------------------------------------------------------------------
// Kernel 2: one workgroup per window (1024 total), 8 waves = 8 heads.
// ---------------------------------------------------------------------------
__global__ __launch_bounds__(256, 1) void win_attn_k(
    const float* __restrict__ x, const float* __restrict__ qkv_b,
    const float* __restrict__ proj_b,
    const h16* __restrict__ wq, const h16* __restrict__ wp,
    float* __restrict__ out) {
  extern __shared__ __align__(16) h16 smem[];

  const int tid  = threadIdx.x;
  const int lane = tid & 31;
  const int wav  = tid >> 5;          // head index 0..7
  const int g    = lane >> 4;         // half-wave group
  const int l16  = lane & 15;
  const int win  = blockIdx.x;        // 0..1023
  const long base = (long)win * (kW * kC);

  h16* xr = smem;                              // x (phase 1-2) then attn-out (3-4)
  h16* qk = smem + XSZ + wav * WSZ;            // this head's q|k rows, later P
  h16* vt = qk + QKSZ;                         // this head's V transposed [32][128]

  // ---- Phase 1: window x -> LDS as f16, padded row-major ----
  {
    const float* xp = x + base;
#pragma unroll
    for (int it = 0; it < 32; ++it) {
      const int e = it * 1024 + tid * 4;
      const float4 v = *(const float4*)(xp + e);
      const int r = e >> 8, c = e & 255;
      v4h h; h[0] = (h16)v.x; h[1] = (h16)v.y; h[2] = (h16)v.z; h[3] = (h16)v.w;
      *(v4h*)(xr + r * XS + c) = h;
    }
  }
  __syncthreads();

  // ---- Phase 2: QKV gemm, per-head slices. nt tiles: q:0-15 k:16-31 v:32-47 ----
  for (int ntg = 0; ntg < 3; ++ntg) {
    for (int sub = 0; sub < 2; ++sub) {
      const int nt = ntg * 16 + wav * 2 + sub;
      v16h bfr[kKT];
#pragma unroll
      for (int kt = 0; kt < kKT; ++kt) {
        const h16* p = wq + ((size_t)((kt * kNTqkv + nt) * 32 + lane)) * 16;
        bfr[kt] = ld_frag2(p, p + 8);
      }
      const float bias = qkv_b[nt * 16 + l16];
      for (int mt = 0; mt < 8; ++mt) {
        v8f acc;
#pragma unroll
        for (int i = 0; i < 8; ++i) acc[i] = bias;
        const int row = mt * 16 + l16;
#pragma unroll
        for (int kt = 0; kt < kKT; ++kt) {
          const h16* p = xr + row * XS + kt * 32 + g * 16;
          const v16h a = ld_frag2(p, p + 8);
          acc = __builtin_amdgcn_wmma_f32_16x16x32_f16(
              false, a, false, bfr[kt], (short)0, acc, false, false);
        }
        if (ntg < 2) {                 // q or k: row-major per-head slice
          const int col = ntg * 32 + sub * 16 + l16;
#pragma unroll
          for (int i = 0; i < 8; ++i)
            qk[(mt * 16 + g * 8 + i) * QKS + col] = (h16)acc[i];
        } else {                       // v: store transposed, packed b128
          const int vc = sub * 16 + l16;
          v8h pk;
#pragma unroll
          for (int i = 0; i < 8; ++i) pk[i] = (h16)acc[i];
          *(v8h*)(vt + vc * VTS + mt * 16 + g * 8) = pk;
        }
      }
    }
  }
  __syncthreads();

  // ---- Phase 3: attention for this head ----
  v16h qa[8];                                   // Q A-frags [16x32], one per mt
#pragma unroll
  for (int mt = 0; mt < 8; ++mt) {
    const h16* p = qk + (mt * 16 + l16) * QKS + g * 16;
    qa[mt] = ld_frag2(p, p + 8);
  }
  v16h kb[8];                                   // K^T B-frags [32x16]
#pragma unroll
  for (int nt = 0; nt < 8; ++nt) {
    const h16* p = qk + (nt * 16 + l16) * QKS + 32 + g * 16;
    kb[nt] = ld_frag2(p, p + 8);
  }
  v16h vb[4][2];                                // V B-frags [32x16] from v^T
#pragma unroll
  for (int kt = 0; kt < 4; ++kt)
#pragma unroll
    for (int vtl = 0; vtl < 2; ++vtl) {
      const h16* p = vt + (vtl * 16 + l16) * VTS + kt * 32 + g * 16;
      vb[kt][vtl] = ld_frag2(p, p + 8);
    }

  const float SC = 0.17677669529663687f * 1.4426950408889634f; // hd^-0.5 * log2(e)

#pragma unroll
  for (int mt = 0; mt < 8; ++mt) {
    v8f s[8];
#pragma unroll
    for (int nt = 0; nt < 8; ++nt) {
      v8f z;
#pragma unroll
      for (int i = 0; i < 8; ++i) z[i] = 0.f;
      s[nt] = __builtin_amdgcn_wmma_f32_16x16x32_f16(
          false, qa[mt], false, kb[nt], (short)0, z, false, false);
    }
    // row-wise softmax: D layout => row M=g*8+i lives in VGPR i across 16 lanes
    float mx[8], sm[8];
#pragma unroll
    for (int i = 0; i < 8; ++i) {
      float v = s[0][i];
#pragma unroll
      for (int nt = 1; nt < 8; ++nt) v = fmaxf(v, s[nt][i]);
      mx[i] = v;
    }
#pragma unroll
    for (int d = 1; d < 16; d <<= 1)
#pragma unroll
      for (int i = 0; i < 8; ++i) mx[i] = fmaxf(mx[i], __shfl_xor(mx[i], d, 32));
#pragma unroll
    for (int i = 0; i < 8; ++i) sm[i] = 0.f;
#pragma unroll
    for (int nt = 0; nt < 8; ++nt)
#pragma unroll
      for (int i = 0; i < 8; ++i) {
        const float e = exp2f((s[nt][i] - mx[i]) * SC);
        s[nt][i] = e; sm[i] += e;
      }
#pragma unroll
    for (int d = 1; d < 16; d <<= 1)
#pragma unroll
      for (int i = 0; i < 8; ++i) sm[i] += __shfl_xor(sm[i], d, 32);
    float inv[8];
#pragma unroll
    for (int i = 0; i < 8; ++i) inv[i] = 1.0f / sm[i];
    // P -> per-wave LDS scratch (reuses q/k region; Q/K already in regs)
#pragma unroll
    for (int nt = 0; nt < 8; ++nt)
#pragma unroll
      for (int i = 0; i < 8; ++i)
        qk[(g * 8 + i) * VTS + nt * 16 + l16] = (h16)(s[nt][i] * inv[i]);
    // reload P as A-frags (in-wave LDS ordering guarantees visibility)
    v16h pa[4];
#pragma unroll
    for (int kt = 0; kt < 4; ++kt) {
      const h16* p = qk + l16 * VTS + kt * 32 + g * 16;
      pa[kt] = ld_frag2(p, p + 8);
    }
    // out tile [16 x 32] = P @ V, to out region (aliases x region)
#pragma unroll
    for (int vtl = 0; vtl < 2; ++vtl) {
      v8f o;
#pragma unroll
      for (int i = 0; i < 8; ++i) o[i] = 0.f;
#pragma unroll
      for (int kt = 0; kt < 4; ++kt)
        o = __builtin_amdgcn_wmma_f32_16x16x32_f16(
            false, pa[kt], false, vb[kt][vtl], (short)0, o, false, false);
#pragma unroll
      for (int i = 0; i < 8; ++i)
        xr[(mt * 16 + g * 8 + i) * XS + wav * 32 + vtl * 16 + l16] = (h16)o[i];
    }
  }
  __syncthreads();

  // ---- Phase 4: projection gemm, f32 out + bias ----
  for (int sub = 0; sub < 2; ++sub) {
    const int nt = wav * 2 + sub;
    v16h bfr[kKT];
#pragma unroll
    for (int kt = 0; kt < kKT; ++kt) {
      const h16* p = wp + ((size_t)((kt * kNTproj + nt) * 32 + lane)) * 16;
      bfr[kt] = ld_frag2(p, p + 8);
    }
    const float bias = proj_b[nt * 16 + l16];
    for (int mt = 0; mt < 8; ++mt) {
      v8f acc;
#pragma unroll
      for (int i = 0; i < 8; ++i) acc[i] = bias;
      const int row = mt * 16 + l16;
#pragma unroll
      for (int kt = 0; kt < kKT; ++kt) {
        const h16* p = xr + row * XS + kt * 32 + g * 16;
        const v16h a = ld_frag2(p, p + 8);
        acc = __builtin_amdgcn_wmma_f32_16x16x32_f16(
            false, a, false, bfr[kt], (short)0, acc, false, false);
      }
      float* op = out + ((long)win * kW + mt * 16 + g * 8) * kC + nt * 16 + l16;
#pragma unroll
      for (int i = 0; i < 8; ++i) op[(long)i * kC] = acc[i];
    }
  }
}

extern "C" void kernel_launch(void* const* d_in, const int* in_sizes, int n_in,
                              void* d_out, int out_size, void* d_ws, size_t ws_size,
                              hipStream_t stream) {
  (void)in_sizes; (void)n_in; (void)out_size; (void)ws_size;
  const float* x      = (const float*)d_in[0];
  const float* qkv_w  = (const float*)d_in[1];
  const float* qkv_b  = (const float*)d_in[2];
  const float* proj_w = (const float*)d_in[3];
  const float* proj_b = (const float*)d_in[4];
  float* out = (float*)d_out;

  h16* wq = (h16*)d_ws;                          // 196608 halves
  h16* wp = wq + (size_t)kKT * kNTqkv * 32 * 16; // +65536 halves (512KB total)

  pack_weights_k<<<64, 256, 0, stream>>>(qkv_w, proj_w, wq, wp);

  const size_t shmem = (size_t)LDS_HALVES * sizeof(h16);  // 284672 B
  win_attn_k<<<1024, 256, shmem, stream>>>(x, qkv_b, proj_b, wq, wp, out);
}
